// KBABlock_14259291423210
// MI455X (gfx1250) — matrix operand order repaired and checked by reading, not compiled, not run on hardware
//
#include <hip/hip_runtime.h>
#include <hip/hip_bf16.h>
#include <math.h>

// ---------------- problem constants ----------------
#define Bn    2
#define DIM   48
#define Hh    128
#define Ww    128
#define Pp    (Hh*Ww)        // 16384
#define NPIX  (Bn*Pp)        // 32768
#define HID   96
#define NSET  32
#define NTILE (NPIX/16)      // 2048 pixel tiles of 16

typedef __attribute__((ext_vector_type(16))) __bf16 v16bf;
typedef __attribute__((ext_vector_type(8)))  float  v8f;

union ABfrag { v16bf v; uint4 q[2]; };

__device__ __forceinline__ unsigned short f2bf(float f) {
    union { float f; unsigned u; } v; v.f = f;
    unsigned r = v.u + 0x7FFFu + ((v.u >> 16) & 1u);   // round-to-nearest-even
    return (unsigned short)(r >> 16);
}
__device__ __forceinline__ float gelu_exact(float x) {
    return 0.5f * x * (1.0f + erff(x * 0.70710678118654752440f));
}

// ---------------------------------------------------------------------------
// K0: pack WMMA B matrices (bf16).
//  Bw layout: [d(0..36)][c(0..95)][n(0..31)], d<36 from kba_w (f = c*36+d),
//             d==36 is the bias column block from kba_b.
//  Bpo layout: [kc(0..2)][co(0..47)][nn(0..31)] = po_w[co][kc*32+nn].
// ---------------------------------------------------------------------------
__global__ void k0_pack(const float* __restrict__ kba_w, const float* __restrict__ kba_b,
                        const float* __restrict__ po_w,
                        unsigned short* __restrict__ Bw, unsigned short* __restrict__ Bpo) {
    int i = blockIdx.x * blockDim.x + threadIdx.x;
    const int NB = 37 * 96 * 32;
    if (i < NB) {
        int n = i & 31;
        int c = (i >> 5) % 96;
        int d = i / (96 * 32);
        float v = (d < 36) ? kba_w[n * 3456 + c * 36 + d] : kba_b[n * 96 + c];
        Bw[i] = f2bf(v);
    } else if (i < NB + 3 * 48 * 32) {
        int j  = i - NB;
        int nn = j & 31;
        int co = (j >> 5) % 48;
        int kc = j / (48 * 32);
        Bpo[j] = f2bf(po_w[co * 96 + kc * 32 + nn]);
    }
}

// ---------------------------------------------------------------------------
// K1: per-pixel 1x1 convs: y1 = dw1(x), ya = c1a(x), attlin = c211(x).
// ---------------------------------------------------------------------------
__global__ void k1_pointwise(const float* __restrict__ x,
                             const float* __restrict__ dw1_w, const float* __restrict__ dw1_b,
                             const float* __restrict__ c1a_w, const float* __restrict__ c1a_b,
                             const float* __restrict__ c211_w, const float* __restrict__ c211_b,
                             float* __restrict__ y1, float* __restrict__ ya,
                             float* __restrict__ attlin) {
    int t = blockIdx.x * blockDim.x + threadIdx.x;
    if (t >= NPIX) return;
    int b = t / Pp, p = t % Pp;
    float xv[DIM];
    #pragma unroll
    for (int ci = 0; ci < DIM; ++ci) xv[ci] = x[(size_t)(b * DIM + ci) * Pp + p];
    for (int c = 0; c < HID; ++c) {
        float a1 = dw1_b[c], a2 = c1a_b[c];
        #pragma unroll 8
        for (int ci = 0; ci < DIM; ++ci) {
            a1 = fmaf(dw1_w[c * DIM + ci], xv[ci], a1);
            a2 = fmaf(c1a_w[c * DIM + ci], xv[ci], a2);
        }
        y1[(size_t)(b * HID + c) * Pp + p] = a1;
        ya[(size_t)(b * HID + c) * Pp + p] = a2;
    }
    for (int n = 0; n < NSET; ++n) {
        float a = c211_b[n];
        #pragma unroll 8
        for (int ci = 0; ci < DIM; ++ci) a = fmaf(c211_w[n * DIM + ci], xv[ci], a);
        attlin[(size_t)t * NSET + n] = a;
    }
}

// ---------------------------------------------------------------------------
// K2: spatial 3x3 stage: gelu(dw2(y1)), uf = c1b(ya), attention coefficients
//     att = c2b(SimpleGate(c2a(x)))*gamma + attlin  -> bf16 [p][32].
// ---------------------------------------------------------------------------
__global__ void k2_spatial(const float* __restrict__ x,
                           const float* __restrict__ y1, const float* __restrict__ ya,
                           const float* __restrict__ dw2_w, const float* __restrict__ dw2_b,
                           const float* __restrict__ c1b_w, const float* __restrict__ c1b_b,
                           const float* __restrict__ c2a_w, const float* __restrict__ c2a_b,
                           const float* __restrict__ c2b_w, const float* __restrict__ c2b_b,
                           const float* __restrict__ attgamma, const float* __restrict__ attlin,
                           float* __restrict__ gelu_x1, float* __restrict__ uf,
                           unsigned short* __restrict__ att_bf) {
    int t = blockIdx.x * blockDim.x + threadIdx.x;
    if (t >= NPIX) return;
    int b = t / Pp, p = t % Pp;
    int y0 = p >> 7, x0 = p & 127;

    int  nidx[9];
    bool nok[9];
    #pragma unroll
    for (int kk = 0; kk < 9; ++kk) {
        int gy = y0 + kk / 3 - 1, gx = x0 + kk % 3 - 1;
        nok[kk]  = ((unsigned)gy < (unsigned)Hh) && ((unsigned)gx < (unsigned)Ww);
        nidx[kk] = gy * Ww + gx;
    }

    // depthwise 3x3 for both branches
    for (int c = 0; c < HID; ++c) {
        float a1 = dw2_b[c], a2 = c1b_b[c];
        const float* b1 = y1 + (size_t)(b * HID + c) * Pp;
        const float* b2 = ya + (size_t)(b * HID + c) * Pp;
        #pragma unroll
        for (int kk = 0; kk < 9; ++kk) {
            if (nok[kk]) {
                a1 = fmaf(dw2_w[c * 9 + kk], b1[nidx[kk]], a1);
                a2 = fmaf(c1b_w[c * 9 + kk], b2[nidx[kk]], a2);
            }
        }
        gelu_x1[(size_t)t * HID + c]        = gelu_exact(a1);
        uf[(size_t)(b * HID + c) * Pp + p]  = a2;
    }

    // grouped 3x3 gate conv (24 groups, 2 in-ch each) + SimpleGate
    float gt[24];
    for (int j = 0; j < 24; ++j) {
        float acc = c2a_b[j];
        #pragma unroll
        for (int q = 0; q < 2; ++q) {
            const float* bx = x + (size_t)(b * DIM + 2 * j + q) * Pp;
            #pragma unroll
            for (int kk = 0; kk < 9; ++kk)
                if (nok[kk]) acc = fmaf(c2a_w[(j * 2 + q) * 9 + kk], bx[nidx[kk]], acc);
        }
        gt[j] = acc;
    }
    float sg[12];
    #pragma unroll
    for (int jj = 0; jj < 12; ++jj) sg[jj] = gt[jj] * gt[jj + 12];

    for (int n = 0; n < NSET; ++n) {
        float s = c2b_b[n];
        #pragma unroll
        for (int jj = 0; jj < 12; ++jj) s = fmaf(c2b_w[n * 12 + jj], sg[jj], s);
        float a = s * attgamma[n] + attlin[(size_t)t * NSET + n];
        att_bf[(size_t)t * NSET + n] = f2bf(a);
    }
}

// ---------------------------------------------------------------------------
// K3: fused KBA core. Per block: 64 pixels of one image row, 4 waves.
//     Double-buffered async staging: while slab u is consumed by 37
//     v_wmma_f32_16x16x32_bf16 per wave, slab u+1 streams into the other
//     LDS buffer via global_load_async_to_lds_b32 (ASYNCcnt-tracked).
//     Every wave issues exactly 25 async instructions per slab (full EXEC;
//     OOB/tail lanes are routed to a dummy LDS slot with a clamped valid
//     address), so "s_wait_asynccnt <= 25" after issuing slab u+1 proves
//     slab u is complete (async loads retire in order).
// ---------------------------------------------------------------------------
#define SLAB_ELEMS (16*3*66)   // 3168 real elements per slab
#define SLAB_ITERS 25          // ceil(3200/128): uniform per-thread issue count

__global__ void __launch_bounds__(128) k3_kba(const unsigned short* __restrict__ att_bf,
                                              const unsigned short* __restrict__ Bw,
                                              const float* __restrict__ uf,
                                              const float* __restrict__ gelu_x1,
                                              const float* __restrict__ ga1,
                                              unsigned short* __restrict__ zout) {
    __shared__ float smb[2][16 * 3 * 67];   // [buf][chL][ry][cx], padded stride 67
    __shared__ float smdummy[128];          // per-thread async dump slot

    int bid = blockIdx.x;               // 512 blocks
    int b   = bid >> 8;
    int pix = (bid & 255) * 64;
    int y0  = pix >> 7;
    int x0  = pix & 127;                // 0 or 64
    int tid  = threadIdx.x;
    int lane = tid & 31;
    int w    = tid >> 5;
    bool lo  = lane < 16;
    int  col = lane & 15;
    int  m0  = lo ? 0 : 8;
    size_t bp0 = (size_t)b * Pp + (size_t)y0 * Ww + x0 + 16 * w;

    const uint32_t dummy_laddr = (uint32_t)(uintptr_t)(&smdummy[tid]);

    // A fragment: 16 pixels x 32 basis coefficients (bf16, contiguous k = 16h + s)
    ABfrag ua;
    {
        const unsigned short* ap = att_bf + (bp0 + col) * NSET;
        ua.q[0] = *(const uint4*)(ap + (lo ? 0 : 16));
        ua.q[1] = *(const uint4*)(ap + (lo ? 8 : 24));
    }
    v16bf afrag = ua.v;
    const v8f vzero = {0, 0, 0, 0, 0, 0, 0, 0};

    // issue the async halo copy for channel slab `u` into buffer u&1
    auto issue_slab = [&](int u) {
        float* sb = &smb[u & 1][0];
        const uint32_t sb_laddr = (uint32_t)(uintptr_t)sb;
        #pragma unroll 1
        for (int it = 0; it < SLAB_ITERS; ++it) {
            int i   = tid + it * 128;
            int chL = i / 198, rem = i % 198;
            int ry  = rem / 66, cx = rem % 66;
            int gy  = y0 + ry - 1, gx = x0 - 1 + cx;
            bool ok = (i < SLAB_ELEMS) &&
                      ((unsigned)gy < (unsigned)Hh) && ((unsigned)gx < (unsigned)Ww);
            int lidx = chL * 201 + ry * 67 + cx;
            uint32_t la = ok ? (sb_laddr + (uint32_t)lidx * 4u) : dummy_laddr;
            const float* gp = ok
                ? (uf + (size_t)(b * HID + 16 * u + chL) * Pp + (size_t)(gy * Ww + gx))
                : uf;                                  // clamped, always valid
            asm volatile("global_load_async_to_lds_b32 %0, %1, off"
                         :: "v"(la), "v"(gp) : "memory");
            if (!ok && i < SLAB_ELEMS) sb[lidx] = 0.f;  // zero-pad halo (disjoint addrs)
        }
    };

    issue_slab(0);

    for (int u = 0; u < 6; ++u) {
        if (u < 5) {
            issue_slab(u + 1);
            // <=25 outstanding == the just-issued batch only -> slab u landed
            asm volatile("s_wait_asynccnt 0x19" ::: "memory");
        } else {
            asm volatile("s_wait_asynccnt 0x0" ::: "memory");
        }
        __syncthreads();                 // slab u visible to all waves

        const float* sb = &smb[u & 1][0];

        // bias tile (d = 36)
        ABfrag ub;
        {
            const unsigned short* bp_ = Bw + ((size_t)(36 * 96 + 16 * u + col)) * 32 + (lo ? 0 : 16);
            ub.q[0] = ((const uint4*)bp_)[0];
            ub.q[1] = ((const uint4*)bp_)[1];
        }
        v8f acc = __builtin_amdgcn_wmma_f32_16x16x32_bf16(false, afrag, false, ub.v,
                                                          (short)0, vzero, false, false);

        for (int d = 0; d < 36; ++d) {
            const unsigned short* bp_ = Bw + ((size_t)(d * 96 + 16 * u + col)) * 32 + (lo ? 0 : 16);
            __builtin_prefetch(bp_ + 96 * 32, 0, 3);    // next tap's B tile
            ABfrag ubd;
            ubd.q[0] = ((const uint4*)bp_)[0];
            ubd.q[1] = ((const uint4*)bp_)[1];
            v8f tt = __builtin_amdgcn_wmma_f32_16x16x32_bf16(false, afrag, false, ubd.v,
                                                             (short)0, vzero, false, false);
            int ci = d / 9, kk = d % 9;
            int dy = kk / 3, dx = kk % 3;
            int chL = ((col >> 2) << 2) + ci;           // in-slab input channel
            const float* base = &sb[chL * 201 + dy * 67 + 16 * w + dx + m0];
            #pragma unroll
            for (int r = 0; r < 8; ++r) acc[r] = fmaf(tt[r], base[r], acc[r]);
        }

        int c = 16 * u + col;
        float ga = ga1[c];
        const float* cbase = &sb[col * 201 + 67 + 16 * w + 1 + m0];  // uf center
        #pragma unroll
        for (int r = 0; r < 8; ++r) {
            int m = m0 + r;
            float x2 = fmaf(acc[r], ga, cbase[r]);
            float g1 = gelu_x1[(bp0 + m) * HID + c];
            zout[(bp0 + m) * HID + c] = f2bf(g1 * x2);
        }
        __syncthreads();                 // all readers done before buffer reuse
    }
}

// ---------------------------------------------------------------------------
// K4: output 1x1 conv as WMMA GEMM: out[p,co] = z[p,:96] . po_w[co,:96] + po_b.
//     K = 96 -> 3 chained WMMA accumulations; N = 48 -> 3 column tiles.
// ---------------------------------------------------------------------------
__global__ void __launch_bounds__(256) k4_out(const unsigned short* __restrict__ z,
                                              const unsigned short* __restrict__ Bpo,
                                              const float* __restrict__ po_b,
                                              float* __restrict__ out) {
    int wave = (blockIdx.x * blockDim.x + threadIdx.x) >> 5;   // 0..2047
    int lane = threadIdx.x & 31;
    bool lo  = lane < 16;
    int  col = lane & 15;
    int  m0  = lo ? 0 : 8;
    size_t bp0 = (size_t)wave * 16;
    int b    = (int)(bp0 / Pp);
    int prow = (int)(bp0 % Pp);

    v16bf a[3];
    #pragma unroll
    for (int kc = 0; kc < 3; ++kc) {
        ABfrag ua;
        const unsigned short* ap = z + (bp0 + col) * HID + kc * 32;
        ua.q[0] = *(const uint4*)(ap + (lo ? 0 : 16));
        ua.q[1] = *(const uint4*)(ap + (lo ? 8 : 24));
        a[kc] = ua.v;
    }
    #pragma unroll
    for (int cot = 0; cot < 3; ++cot) {
        v8f acc = {0, 0, 0, 0, 0, 0, 0, 0};
        #pragma unroll
        for (int kc = 0; kc < 3; ++kc) {
            ABfrag ub;
            const unsigned short* bp_ = Bpo + ((size_t)(kc * 48 + cot * 16 + col)) * 32 + (lo ? 0 : 16);
            ub.q[0] = ((const uint4*)bp_)[0];
            ub.q[1] = ((const uint4*)bp_)[1];
            acc = __builtin_amdgcn_wmma_f32_16x16x32_bf16(false, a[kc], false, ub.v,
                                                          (short)0, acc, false, false);
        }
        int co = cot * 16 + col;
        float bias = po_b[co];
        #pragma unroll
        for (int r = 0; r < 8; ++r)
            out[(size_t)(b * DIM + co) * Pp + prow + m0 + r] = acc[r] + bias;
    }
}

// ---------------------------------------------------------------------------
extern "C" void kernel_launch(void* const* d_in, const int* in_sizes, int n_in,
                              void* d_out, int out_size, void* d_ws, size_t ws_size,
                              hipStream_t stream) {
    (void)in_sizes; (void)n_in; (void)out_size; (void)ws_size;
    const float* x      = (const float*)d_in[0];
    const float* dw1_w  = (const float*)d_in[1];
    const float* dw1_b  = (const float*)d_in[2];
    const float* dw2_w  = (const float*)d_in[3];
    const float* dw2_b  = (const float*)d_in[4];
    const float* c1a_w  = (const float*)d_in[5];
    const float* c1a_b  = (const float*)d_in[6];
    const float* c1b_w  = (const float*)d_in[7];
    const float* c1b_b  = (const float*)d_in[8];
    const float* c2a_w  = (const float*)d_in[9];
    const float* c2a_b  = (const float*)d_in[10];
    const float* c2b_w  = (const float*)d_in[11];
    const float* c2b_b  = (const float*)d_in[12];
    const float* c211_w = (const float*)d_in[13];
    const float* c211_b = (const float*)d_in[14];
    const float* attg   = (const float*)d_in[15];
    const float* ga1    = (const float*)d_in[16];
    const float* kba_w  = (const float*)d_in[17];
    const float* kba_b  = (const float*)d_in[18];
    const float* po_w   = (const float*)d_in[19];
    const float* po_b   = (const float*)d_in[20];
    float* out = (float*)d_out;

    // workspace carve-up (256B aligned regions)
    char* ws = (char*)d_ws;
    size_t off = 0;
    auto carve = [&](size_t bytes) -> char* {
        char* p = ws + off;
        off += (bytes + 255) & ~((size_t)255);
        return p;
    };
    float*          y1      = (float*)carve((size_t)NPIX * HID * 4);
    float*          ya      = (float*)carve((size_t)NPIX * HID * 4);
    float*          uf      = (float*)carve((size_t)NPIX * HID * 4);
    float*          gx1     = (float*)carve((size_t)NPIX * HID * 4);
    float*          attlin  = (float*)carve((size_t)NPIX * NSET * 4);
    unsigned short* att_bf  = (unsigned short*)carve((size_t)NPIX * NSET * 2);
    unsigned short* zbuf    = (unsigned short*)carve((size_t)NPIX * HID * 2);
    unsigned short* Bw      = (unsigned short*)carve((size_t)37 * 96 * 32 * 2);
    unsigned short* Bpo     = (unsigned short*)carve((size_t)3 * 48 * 32 * 2);

    {   // K0: pack B matrices
        int total = 37 * 96 * 32 + 3 * 48 * 32;
        k0_pack<<<(total + 255) / 256, 256, 0, stream>>>(kba_w, kba_b, po_w, Bw, Bpo);
    }
    {   // K1: pointwise stage
        k1_pointwise<<<(NPIX + 255) / 256, 256, 0, stream>>>(
            x, dw1_w, dw1_b, c1a_w, c1a_b, c211_w, c211_b, y1, ya, attlin);
    }
    {   // K2: spatial stage + attention
        k2_spatial<<<(NPIX + 255) / 256, 256, 0, stream>>>(
            x, y1, ya, dw2_w, dw2_b, c1b_w, c1b_b, c2a_w, c2a_b,
            c2b_w, c2b_b, attg, attlin, gx1, uf, att_bf);
    }
    {   // K3: fused KBA WMMA core (512 blocks x 128 threads = 64 px/block)
        k3_kba<<<NPIX / 64, 128, 0, stream>>>(att_bf, Bw, uf, gx1, ga1, zbuf);
    }
    {   // K4: output 1x1 conv GEMM (2048 wave-tiles, 8 waves/block)
        k4_out<<<NTILE / 8, 256, 0, stream>>>(zbuf, Bpo, po_b, out);
    }
}
